// StackedRNN_22368189677861
// MI455X (gfx1250) — matrix operand order repaired and checked by reading, not compiled, not run on hardware
//
#include <hip/hip_runtime.h>
#include <hip/hip_bf16.h>

// ---------------------------------------------------------------------------
// StackedRNN for MI455X (gfx1250): bf16 WMMA everywhere, register-resident
// recurrent weights in the GRU scan (batch-parallel, no inter-WG sync),
// async global->LDS slab prefetch (ASYNCcnt) for the recurrence inputs.
// B=64 T=2048 IN=32 H=256 S=256 OUT=32 NB=4
// ---------------------------------------------------------------------------

typedef __bf16 bf16;
typedef __attribute__((ext_vector_type(16))) __bf16 bf16x16;
typedef __attribute__((ext_vector_type(8)))  float  f32x8;

#define BT   131072      // B*T
#define HD   256
#define SD   256
#define G3   768         // 3*S
#define TT   2048
#define BB   64

union Frag { bf16x16 v; uint4 q[2]; };

__device__ __forceinline__ f32x8 fzero() {
    f32x8 z;
#pragma unroll
    for (int i = 0; i < 8; ++i) z[i] = 0.0f;
    return z;
}

// A-matrix 16x32 bf16 fragment from row-major [m][k] buffer (stride ld elems).
// ISA layout: lane l: M=l&15; VGPR0-3 hold K=(l>>4)*8 .. +7, VGPR4-7 hold +16.
__device__ __forceinline__ Frag load_a(const bf16* base, int ld) {
    const int l = threadIdx.x & 31, r = l & 15, hi = l >> 4;
    const bf16* p = base + (size_t)r * ld + hi * 8;
    Frag f;
    f.q[0] = *(const uint4*)(p);
    f.q[1] = *(const uint4*)(p + 16);
    return f;
}

// B-matrix 32x16 bf16 fragment from row-major weight [n][k] buffer.
// ISA layout: lane l: N=l&15; VGPR0-3 hold K=(l>>4)*8 .. +7, VGPR4-7 hold +16.
__device__ __forceinline__ Frag load_b(const bf16* base, int ld) {
    const int l = threadIdx.x & 31, n = l & 15, hi = l >> 4;
    const bf16* p = base + (size_t)n * ld + hi * 8;
    Frag f;
    f.q[0] = *(const uint4*)(p);
    f.q[1] = *(const uint4*)(p + 16);
    return f;
}

__device__ __forceinline__ f32x8 wmma_bf16(Frag a, Frag b, f32x8 c) {
    return __builtin_amdgcn_wmma_f32_16x16x32_bf16(false, a.v, false, b.v,
                                                   (short)0, c, false, false);
}

__device__ __forceinline__ float sigmoidf_(float x) {
    return 1.0f / (1.0f + __expf(-x));
}

// --- CDNA5 async global->LDS (tracked with ASYNCcnt) -----------------------
__device__ __forceinline__ void async_load_b128(unsigned lds_byte_off,
                                                const void* gaddr) {
    asm volatile("global_load_async_to_lds_b128 %0, %1, off"
                 :: "v"(lds_byte_off), "v"(gaddr)
                 : "memory");
}
__device__ __forceinline__ void wait_async0() {
    asm volatile("s_wait_asynccnt 0x0" ::: "memory");
}

// --------------------------------- misc ------------------------------------

__global__ void cvt_bf16(const float* __restrict__ src, bf16* __restrict__ dst,
                         size_t n) {
    size_t i = (size_t)blockIdx.x * blockDim.x + threadIdx.x;
    const size_t stride = (size_t)gridDim.x * blockDim.x;
    for (; i < n; i += stride) dst[i] = (bf16)src[i];
}

// ------------------------------- encoder -----------------------------------
// h(BT,256) = x(BT,32) @ enc_w(256,32)^T + enc_b ; one WMMA K-step (K=32).
__global__ void enc_gemm(const bf16* __restrict__ X, const bf16* __restrict__ W,
                         const float* __restrict__ bias, float* __restrict__ H) {
    const int mtile = blockIdx.x;                  // 8192 tiles of 16 rows
    const int wave = threadIdx.x >> 5;
    const int l = threadIdx.x & 31, hi = l >> 4, col = l & 15;
    Frag a = load_a(X + (size_t)mtile * 16 * 32, 32);
#pragma unroll
    for (int nt = 0; nt < 2; ++nt) {
        const int n0 = (wave * 2 + nt) * 16;
        Frag b = load_b(W + (size_t)n0 * 32, 32);
        f32x8 acc = fzero();
        acc = wmma_bf16(a, b, acc);
        const float bb = bias[n0 + col];
        float* out = H + (size_t)mtile * 16 * HD + n0 + col;
#pragma unroll
        for (int v = 0; v < 8; ++v)
            out[(size_t)(v + 8 * hi) * HD] = acc[v] + bb;
    }
}

// ------------------------------ batch norm ---------------------------------
// Deterministic two-pass reduction: 1024 partials per channel, then serial sum.
__global__ void bn_stats(const float* __restrict__ H, float* __restrict__ part) {
    const int c = threadIdx.x;                     // 256 channels
    const size_t row0 = (size_t)blockIdx.x * 128;  // 1024 blocks * 128 rows
    float s = 0.f, s2 = 0.f;
    for (int r = 0; r < 128; ++r) {
        float v = H[(row0 + r) * HD + c];
        s += v; s2 += v * v;
    }
    part[(size_t)blockIdx.x * 512 + c]       = s;
    part[(size_t)blockIdx.x * 512 + 256 + c] = s2;
}

__global__ void bn_finalize(const float* __restrict__ part,
                            float* __restrict__ mr) {
    const int c = threadIdx.x;
    float s = 0.f, s2 = 0.f;
    for (int k = 0; k < 1024; ++k) {
        s  += part[(size_t)k * 512 + c];
        s2 += part[(size_t)k * 512 + 256 + c];
    }
    const float invN = 1.0f / (float)BT;
    const float mean = s * invN;
    const float var = s2 * invN - mean * mean;
    mr[c]       = mean;
    mr[256 + c] = rsqrtf(var + 1e-5f);
}

// ---------------------------- input-gate GEMM ------------------------------
// ig(T,B,768) f32 = bn(h)(B,T,256) @ wih(768,256)^T + b_ih
// BatchNorm-apply is fused into the LDS A-tile staging (f32 -> bf16).
// M is ordered (T,B): mtile -> t = mtile>>2, batch base b0 = (mtile&3)*16.
__global__ void ig_gemm(const float* __restrict__ H, const float* __restrict__ MR,
                        const bf16* __restrict__ W, const float* __restrict__ bias,
                        float* __restrict__ IG) {
    __shared__ bf16 sA[16 * HD];
    const int mtile = blockIdx.x;                  // 8192
    {
        const int t = mtile >> 2, b0 = (mtile & 3) * 16;
        const int r = threadIdx.x >> 4;            // tile row (batch b0+r)
        const int q = threadIdx.x & 15;            // 16-float chunk
        const float* src = H + ((size_t)(b0 + r) * TT + t) * HD + q * 16;
#pragma unroll
        for (int u = 0; u < 16; ++u) {
            const int c = q * 16 + u;
            sA[r * HD + c] = (bf16)((src[u] - MR[c]) * MR[256 + c]);
        }
    }
    __syncthreads();
    const int wave = threadIdx.x >> 5;             // 8 waves x 6 ntiles = 48
    const int l = threadIdx.x & 31, hi = l >> 4, col = l & 15;
    f32x8 acc[6];
#pragma unroll
    for (int j = 0; j < 6; ++j) acc[j] = fzero();
#pragma unroll
    for (int kk = 0; kk < 8; ++kk) {
        Frag a = load_a(sA + kk * 32, HD);
#pragma unroll
        for (int j = 0; j < 6; ++j) {
            const int n0 = (wave * 6 + j) * 16;
            Frag b = load_b(W + (size_t)n0 * HD + kk * 32, HD);
            acc[j] = wmma_bf16(a, b, acc[j]);
        }
    }
#pragma unroll
    for (int j = 0; j < 6; ++j) {
        const int n0 = (wave * 6 + j) * 16;
        const float bb = bias[n0 + col];
        float* out = IG + (size_t)mtile * 16 * G3 + n0 + col;
#pragma unroll
        for (int v = 0; v < 8; ++v)
            out[(size_t)(v + 8 * hi) * G3] = acc[j][v] + bb;
    }
}

// -------------------------------- GRU scan ---------------------------------
// 4 independent WGs (16 batch rows each, no inter-WG sync). 16 waves; wave w
// owns channel tile [16w,16w+16) and holds whh fragments for all 3 gates and
// all 8 K-tiles in VGPRs (24 frags * 8 VGPR = 192 VGPRs). Hidden state ping-
// pongs through double-buffered LDS (one barrier/step). The 48KB input-gate
// slab for step t+1 is prefetched with async global->LDS b128 transfers and
// drained with s_wait_asynccnt, overlapping the ASYNC engine with the WMMAs.
__global__ void __launch_bounds__(512) gru_scan(
    const float* __restrict__ IG,   // (T,64,768)
    const bf16* __restrict__ WHH,   // (768,256) bf16
    const float* __restrict__ BN,   // b_n (256)
    bf16* __restrict__ SG)          // gelu(states) (B,T,256) bf16
{
    __shared__ bf16 sh[2][16 * HD];                        // 16 KB
    __shared__ __align__(16) float sIG[2][16 * G3];        // 96 KB
    const int gx = blockIdx.x;                     // batch group 0..3
    const int wave = threadIdx.x >> 5;             // channel tile 0..15
    const int l = threadIdx.x & 31, hi = l >> 4, col = l & 15;
    const int j = wave * 16 + col;                 // this lane's channel

    Frag Wr[8], Wz[8], Wn[8];
#pragma unroll
    for (int kk = 0; kk < 8; ++kk) {
        Wr[kk] = load_b(WHH + ((size_t)(0 * SD + wave * 16)) * HD + kk * 32, HD);
        Wz[kk] = load_b(WHH + ((size_t)(1 * SD + wave * 16)) * HD + kk * 32, HD);
        Wn[kk] = load_b(WHH + ((size_t)(2 * SD + wave * 16)) * HD + kk * 32, HD);
    }
    const float bn_j = BN[j];
    f32x8 hstate = fzero();

    for (int i = threadIdx.x; i < 16 * HD; i += 512) sh[0][i] = (bf16)0.0f;

    // kick off async prefetch of the t=0 input-gate slab
    const float* gslab0 = IG + (size_t)(0 * BB + gx * 16) * G3;
    {
        const unsigned lds0 = (unsigned)(unsigned long long)(const void*)&sIG[0][0];
        const unsigned q = threadIdx.x;
#pragma unroll
        for (int u = 0; u < 6; ++u) {
            const unsigned off = (unsigned)(u * 512 + q) * 16u;
            async_load_b128(lds0 + off, (const char*)gslab0 + off);
        }
    }

    const float c0 = 0.7978845608028654f;  // sqrt(2/pi), gelu tanh approx
    const float c1 = 0.044715f;

#pragma unroll 1
    for (int t = 0; t < TT; ++t) {
        wait_async0();       // my slab-t transfers done
        __syncthreads();     // everyone's transfers + sh[t&1] writes visible

        if (t + 1 < TT) {    // issue slab t+1 into the other buffer
            const float* g = IG + ((size_t)(t + 1) * BB + gx * 16) * G3;
            const unsigned lds0 =
                (unsigned)(unsigned long long)(const void*)&sIG[(t + 1) & 1][0];
            const unsigned q = threadIdx.x;
#pragma unroll
            for (int u = 0; u < 6; ++u) {
                const unsigned off = (unsigned)(u * 512 + q) * 16u;
                async_load_b128(lds0 + off, (const char*)g + off);
            }
        }

        // recurrent matmul: h_prev(16x256) x whh^T(256x[3x16 cols])
        const bf16* shc = sh[t & 1];
        f32x8 aR = fzero(), aZ = fzero(), aN = fzero();
#pragma unroll
        for (int kk = 0; kk < 8; ++kk) {
            Frag a = load_a(shc + kk * 32, HD);
            aR = wmma_bf16(a, Wr[kk], aR);
            aZ = wmma_bf16(a, Wz[kk], aZ);
            aN = wmma_bf16(a, Wn[kk], aN);
        }

        const float* cur = sIG[t & 1];
        bf16* shn = sh[(t + 1) & 1];
#pragma unroll
        for (int v = 0; v < 8; ++v) {
            const int m = v + 8 * hi;
            const float* qp = cur + (size_t)m * G3 + j;
            const float r = sigmoidf_(qp[0] + aR[v]);
            const float z = sigmoidf_(qp[SD] + aZ[v]);
            const float n = tanhf(qp[2 * SD] + r * (aN[v] + bn_j));
            const float hnew = (1.0f - z) * n + z * hstate[v];
            hstate[v] = hnew;
            const float g =
                0.5f * hnew * (1.0f + tanhf(c0 * (hnew + c1 * hnew * hnew * hnew)));
            const int b = gx * 16 + m;
            SG[((size_t)b * TT + t) * SD + j] = (bf16)g;
            shn[m * HD + j] = (bf16)hnew;
        }
    }
}

// --------------------------------- GLU -------------------------------------
// h += (s @ g1w^T + g1b) * sigmoid(s @ g2w^T + g2b), s = gelu(states) bf16.
// For the last block, also emits h as bf16 (decoder input) to skip a pass.
__global__ void glu_gemm(const bf16* __restrict__ S, const bf16* __restrict__ W1,
                         const bf16* __restrict__ W2, const float* __restrict__ B1,
                         const float* __restrict__ B2, float* __restrict__ H,
                         bf16* __restrict__ HB) {
    __shared__ bf16 sA[16 * SD];
    const int mtile = blockIdx.x;                  // 8192
    {
        const uint4* src = (const uint4*)(S + (size_t)mtile * 16 * SD);
        uint4* dst = (uint4*)sA;
        for (int i = threadIdx.x; i < 512; i += 256) dst[i] = src[i];
    }
    __syncthreads();
    const int wave = threadIdx.x >> 5;             // 8 waves x 2 ntiles = 16
    const int l = threadIdx.x & 31, hi = l >> 4, col = l & 15;
    f32x8 a1[2], a2[2];
#pragma unroll
    for (int j = 0; j < 2; ++j) { a1[j] = fzero(); a2[j] = fzero(); }
#pragma unroll
    for (int kk = 0; kk < 8; ++kk) {
        Frag a = load_a(sA + kk * 32, SD);
#pragma unroll
        for (int j = 0; j < 2; ++j) {
            const int n0 = (wave * 2 + j) * 16;
            Frag b1 = load_b(W1 + (size_t)n0 * SD + kk * 32, SD);
            Frag b2 = load_b(W2 + (size_t)n0 * SD + kk * 32, SD);
            a1[j] = wmma_bf16(a, b1, a1[j]);
            a2[j] = wmma_bf16(a, b2, a2[j]);
        }
    }
#pragma unroll
    for (int j = 0; j < 2; ++j) {
        const int n0 = (wave * 2 + j) * 16;
        const float bb1 = B1[n0 + col], bb2 = B2[n0 + col];
        const size_t base = (size_t)mtile * 16 * HD + n0 + col;
#pragma unroll
        for (int v = 0; v < 8; ++v) {
            const size_t off = base + (size_t)(v + 8 * hi) * HD;
            const float va = a1[j][v] + bb1;
            const float vg = a2[j][v] + bb2;
            const float nh = H[off] + va * sigmoidf_(vg);
            H[off] = nh;
            if (HB) HB[off] = (bf16)nh;
        }
    }
}

// ------------------------------- decoder -----------------------------------
// out(BT,32) = tanh(h_bf16(BT,256) @ dec_w(32,256)^T + dec_b)
__global__ void dec_gemm(const bf16* __restrict__ A, const bf16* __restrict__ W,
                         const float* __restrict__ bias, float* __restrict__ OUT) {
    const int wave = threadIdx.x >> 5;
    const int l = threadIdx.x & 31, hi = l >> 4, col = l & 15;
    const size_t mtile = (size_t)blockIdx.x * 8 + wave;   // 1024 blocks
    f32x8 acc[2];
#pragma unroll
    for (int j = 0; j < 2; ++j) acc[j] = fzero();
#pragma unroll
    for (int kk = 0; kk < 8; ++kk) {
        Frag a = load_a(A + mtile * 16 * HD + kk * 32, HD);
#pragma unroll
        for (int j = 0; j < 2; ++j) {
            Frag b = load_b(W + (size_t)j * 16 * HD + kk * 32, HD);
            acc[j] = wmma_bf16(a, b, acc[j]);
        }
    }
#pragma unroll
    for (int j = 0; j < 2; ++j) {
        const float bb = bias[j * 16 + col];
        float* out = OUT + mtile * 16 * 32 + j * 16 + col;
#pragma unroll
        for (int v = 0; v < 8; ++v)
            out[(size_t)(v + 8 * hi) * 32] = tanhf(acc[j][v] + bb);
    }
}

// ------------------------------ host driver --------------------------------

extern "C" void kernel_launch(void* const* d_in, const int* in_sizes, int n_in,
                              void* d_out, int out_size, void* d_ws, size_t ws_size,
                              hipStream_t stream) {
    (void)in_sizes; (void)n_in; (void)out_size; (void)ws_size;
    const float* x     = (const float*)d_in[0];
    const float* enc_w = (const float*)d_in[1];
    const float* enc_b = (const float*)d_in[2];
    const float* wih   = (const float*)d_in[3];
    const float* whh   = (const float*)d_in[4];
    const float* b_ih  = (const float*)d_in[5];
    const float* b_n   = (const float*)d_in[6];
    const float* g1w   = (const float*)d_in[7];
    const float* g1b   = (const float*)d_in[8];
    const float* g2w   = (const float*)d_in[9];
    const float* g2b   = (const float*)d_in[10];
    const float* dec_w = (const float*)d_in[11];
    const float* dec_b = (const float*)d_in[12];
    float* out = (float*)d_out;

    char* p = (char*)d_ws;
    auto alloc = [&](size_t bytes) -> char* {
        char* r = p; p += (bytes + 255) & ~(size_t)255; return r;
    };
    float* h      = (float*)alloc((size_t)BT * HD * 4);
    bf16*  sg     = (bf16*) alloc((size_t)BT * SD * 2);
    bf16*  hbf    = (bf16*) alloc((size_t)BT * HD * 2);
    bf16*  xbf    = (bf16*) alloc((size_t)BT * 32 * 2);
    bf16*  encwbf = (bf16*) alloc((size_t)HD * 32 * 2);
    bf16*  wihbf  = (bf16*) alloc((size_t)4 * G3 * HD * 2);
    bf16*  whhbf  = (bf16*) alloc((size_t)4 * G3 * SD * 2);
    bf16*  g1wbf  = (bf16*) alloc((size_t)4 * HD * SD * 2);
    bf16*  g2wbf  = (bf16*) alloc((size_t)4 * HD * SD * 2);
    bf16*  decwbf = (bf16*) alloc((size_t)32 * HD * 2);
    float* part   = (float*)alloc((size_t)1024 * 512 * 4);
    float* mr     = (float*)alloc(512 * 4);
    float* ig     = (float*)alloc((size_t)BT * G3 * 4);   // (T,B,768)

    // one-time bf16 conversions (amortized; L2-resident afterwards)
    cvt_bf16<<<1024, 256, 0, stream>>>(x, xbf, (size_t)BT * 32);
    cvt_bf16<<<32, 256, 0, stream>>>(enc_w, encwbf, (size_t)HD * 32);
    cvt_bf16<<<512, 256, 0, stream>>>(wih, wihbf, (size_t)4 * G3 * HD);
    cvt_bf16<<<512, 256, 0, stream>>>(whh, whhbf, (size_t)4 * G3 * SD);
    cvt_bf16<<<256, 256, 0, stream>>>(g1w, g1wbf, (size_t)4 * HD * SD);
    cvt_bf16<<<256, 256, 0, stream>>>(g2w, g2wbf, (size_t)4 * HD * SD);
    cvt_bf16<<<8, 256, 0, stream>>>(dec_w, decwbf, (size_t)32 * HD);

    enc_gemm<<<BT / 16, 256, 0, stream>>>(xbf, encwbf, enc_b, h);

    for (int i = 0; i < 4; ++i) {
        bn_stats<<<1024, 256, 0, stream>>>(h, part);
        bn_finalize<<<1, 256, 0, stream>>>(part, mr);
        ig_gemm<<<BT / 16, 256, 0, stream>>>(h, mr, wihbf + (size_t)i * G3 * HD,
                                             b_ih + (size_t)i * G3, ig);
        gru_scan<<<4, 512, 0, stream>>>(ig, whhbf + (size_t)i * G3 * SD,
                                        b_n + (size_t)i * SD, sg);
        glu_gemm<<<BT / 16, 256, 0, stream>>>(sg, g1wbf + (size_t)i * HD * SD,
                                              g2wbf + (size_t)i * HD * SD,
                                              g1b + (size_t)i * HD,
                                              g2b + (size_t)i * HD, h,
                                              (i == 3) ? hbf : (bf16*)nullptr);
    }

    dec_gemm<<<BT / (16 * 8), 256, 0, stream>>>(hbf, decwbf, dec_b, out);
}
